// Mamba2_32306744000747
// MI455X (gfx1250) — matrix operand order repaired and checked by reading, not compiled, not run on hardware
//
#include <hip/hip_runtime.h>
#include <hip/hip_bf16.h>

// ---------------- Problem constants (from reference) ----------------
#define B_      2
#define L_      2048
#define DMODEL_ 2048
#define DIN_    4096      // D_INNER
#define DPROJ_  8512      // D_IN_PROJ
#define CONVD_  4352      // CONV_DIM
#define NH_     64
#define HD_     64
#define DSTATE_ 128
#define NC_     8         // chunks per sequence (L/256)
#define CHUNK_  256

typedef __attribute__((ext_vector_type(16))) __bf16    v16bf;
typedef __attribute__((ext_vector_type(8)))  float     v8f;
typedef __attribute__((ext_vector_type(4)))  unsigned  v4u;
typedef __attribute__((ext_vector_type(8)))  int       v8i;
typedef __attribute__((ext_vector_type(4)))  int       v4i;

union Frag { v16bf v; unsigned u[8]; };

__device__ __forceinline__ unsigned short f2bf(float f) {
  union { float f; unsigned u; } x; x.f = f;
  unsigned r = x.u + 0x7FFFu + ((x.u >> 16) & 1u);   // round-to-nearest-even
  return (unsigned short)(r >> 16);
}
__device__ __forceinline__ unsigned pack2(float a, float b) {
  return (unsigned)f2bf(a) | ((unsigned)f2bf(b) << 16);
}
__device__ __forceinline__ float bf2f(unsigned short s) {
  union { unsigned u; float f; } x; x.u = ((unsigned)s) << 16; return x.f;
}
__device__ __forceinline__ v8f wmma_bf16(const Frag& a, const Frag& b, v8f c) {
  return __builtin_amdgcn_wmma_f32_16x16x32_bf16(false, a.v, false, b.v,
                                                 (short)0, c, false, false);
}

// ---------------- Tensor Data Mover (CDNA5 TDM) ----------------
#if defined(__has_builtin)
#if __has_builtin(__builtin_amdgcn_tensor_load_to_lds) && \
    __has_builtin(__builtin_amdgcn_s_wait_tensorcnt)
#define HAVE_TDM 1
#endif
#endif

#ifdef HAVE_TDM
// 2D tile load: rows x rowElems bf16 elements, global row stride = strideElems.
// LDS receives rows contiguously (rowElems*2 bytes per row).
// This toolchain exposes the 6-arg builtin:
//   (uint32x4 g0, int32x8 g1, int32x4 g2, int32x4 g3, int32x8 g4, i32 cpol)
__device__ __forceinline__ void tdm_load_2d_bf16(
    void* lds_ptr, const void* gptr,
    unsigned rows, unsigned rowElems, unsigned strideElems)
{
  unsigned long long ga = (unsigned long long)gptr;
  unsigned lds_addr = (unsigned)(unsigned long long)lds_ptr;  // LDS offset in [31:0]
  v4u g0;
  g0.x = 1u;                                            // count=1, user mode
  g0.y = lds_addr;                                      // D#.lds_addr
  g0.z = (unsigned)ga;                                  // global_addr[31:0]
  g0.w = (unsigned)((ga >> 32) & 0x01FFFFFFull) | 0x80000000u; // [120:64], type=2
  v8i g1;
  g1[0] = 0x00010000;                                   // data_size=1 (2 bytes)
  g1[1] = (int)((strideElems & 0xFFFFu) << 16);         // tensor_dim0[15:0]
  g1[2] = (int)((strideElems >> 16) | ((rows & 0xFFFFu) << 16)); // dim0 hi | dim1 lo
  g1[3] = (int)((rows >> 16) | (rowElems << 16));       // dim1 hi | tile_dim0
  g1[4] = (int)rows;                                    // tile_dim1 (tile_dim2=0)
  g1[5] = (int)strideElems;                             // tensor_dim0_stride[31:0]
  g1[6] = 0;                                            // stride hi | dim1_stride lo
  g1[7] = 0;
  v4i z4 = {0, 0, 0, 0};
  v8i z8 = {0, 0, 0, 0, 0, 0, 0, 0};
  __builtin_amdgcn_tensor_load_to_lds(g0, g1, z4, z4, z8, 0);
}
#endif

// =====================================================================
// Kernel 1/8: C[M,N] = A[M,K] @ W[N,K]^T, bf16 operands, f32 accum.
// Block tile 128x64, 256 thr = 8 waves (4m x 2n), wave -> 32x32 output
// (2x2 WMMA with fragment reuse). K-tile = 32. All dims divide evenly.
// K-tiles staged in LDS by the Tensor Data Mover when available.
// =====================================================================
__global__ __launch_bounds__(256) void gemm_bf16_wmma(
    const unsigned short* __restrict__ A, const unsigned short* __restrict__ W,
    float* __restrict__ C, int M, int N, int K)
{
  __shared__ unsigned ldsA[128 * 16];   // [m][k/2] bf16 pairs, 8 KB
  __shared__ unsigned ldsW[64 * 16];    // [n][k/2], 4 KB
  const int tid  = threadIdx.x;
  const int lane = tid & 31, wave = tid >> 5;
  const int wm = wave >> 1, wn = wave & 1;
  const int m0 = blockIdx.y * 128, n0 = blockIdx.x * 64;
  const int mlo = lane & 15, kh = lane >> 4;
  const unsigned* A32 = (const unsigned*)A;
  const unsigned* W32 = (const unsigned*)W;
  const int K2 = K >> 1;

  v8f acc[2][2] = {};
  for (int k0 = 0; k0 < K; k0 += 32) {
    __syncthreads();                      // previous tile fully consumed
#ifdef HAVE_TDM
    if (wave == 0) {
      tdm_load_2d_bf16(ldsA, A + (size_t)m0 * K + k0, 128u, 32u, (unsigned)K);
      tdm_load_2d_bf16(ldsW, W + (size_t)n0 * K + k0, 64u, 32u, (unsigned)K);
      __builtin_amdgcn_s_wait_tensorcnt(0);
    }
#else
    __builtin_prefetch(A + (size_t)m0 * K + k0 + 64, 0, 0);
    for (int i = tid; i < 128 * 16; i += 256) {   // raw u32 copy (2 bf16 each)
      int m = i >> 4, kp = i & 15;
      ldsA[i] = A32[(size_t)(m0 + m) * K2 + (k0 >> 1) + kp];
    }
    for (int i = tid; i < 64 * 16; i += 256) {
      int n = i >> 4, kp = i & 15;
      ldsW[i] = W32[(size_t)(n0 + n) * K2 + (k0 >> 1) + kp];
    }
#endif
    __syncthreads();
    Frag a0, a1, b0, b1;
    for (int v = 0; v < 8; ++v) {                 // A-frags (16x32 bf16 layout)
      int k = ((v < 4) ? 0 : 16) + kh * 8 + 2 * (v & 3);
      a0.u[v] = ldsA[(wm * 32 + mlo) * 16 + (k >> 1)];
      a1.u[v] = ldsA[(wm * 32 + 16 + mlo) * 16 + (k >> 1)];
    }
    for (int v = 0; v < 8; ++v) {                 // B-frags (32x16 bf16 layout)
      int k = ((v < 4) ? 0 : 16) + kh * 8 + 2 * (v & 3);
      b0.u[v] = ldsW[(wn * 32 + mlo) * 16 + (k >> 1)];
      b1.u[v] = ldsW[(wn * 32 + 16 + mlo) * 16 + (k >> 1)];
    }
    acc[0][0] = wmma_bf16(a0, b0, acc[0][0]);
    acc[0][1] = wmma_bf16(a0, b1, acc[0][1]);
    acc[1][0] = wmma_bf16(a1, b0, acc[1][0]);
    acc[1][1] = wmma_bf16(a1, b1, acc[1][1]);
  }
  for (int it = 0; it < 2; ++it)
    for (int jt = 0; jt < 2; ++jt)
      for (int r = 0; r < 8; ++r) {               // 16x16 f32 C/D layout
        int row = m0 + wm * 32 + it * 16 + kh * 8 + r;
        int col = n0 + wn * 32 + jt * 16 + mlo;
        C[(size_t)row * N + col] = acc[it][jt][r];
      }
}

// =====================================================================
// Kernel 0: fp32 -> packed bf16 pairs
// =====================================================================
__global__ void cvt_bf16_kernel(const float* __restrict__ src,
                                unsigned* __restrict__ dst, unsigned npairs)
{
  unsigned i = blockIdx.x * blockDim.x + threadIdx.x;
  if (i < npairs) dst[i] = pack2(src[2 * i], src[2 * i + 1]);
}

// =====================================================================
// Kernel 2: causal depthwise conv (width 4) + bias + SiLU, split.
// x kept fp32 (needed for D-skip); B/C emitted directly as bf16.
// =====================================================================
__global__ void conv_silu_kernel(const float* __restrict__ zx,
                                 const float* __restrict__ cw,
                                 const float* __restrict__ cb,
                                 float* __restrict__ x,
                                 unsigned short* __restrict__ Bm_bf,
                                 unsigned short* __restrict__ Cm_bf)
{
  int idx = blockIdx.x * blockDim.x + threadIdx.x;
  if (idx >= B_ * L_ * CONVD_) return;
  int c = idx % CONVD_;
  int l = (idx / CONVD_) % L_;
  int b = idx / (CONVD_ * L_);
  float acc = cb[c];
  for (int w = 0; w < 4; ++w) {
    int ll = l + w - 3;
    if (ll >= 0)
      acc += zx[(size_t)(b * L_ + ll) * DPROJ_ + DIN_ + c] * cw[c * 4 + w];
  }
  float s = acc / (1.f + __expf(-acc));           // SiLU
  size_t row = (size_t)(b * L_ + l);
  if (c < DIN_)              x[row * DIN_ + c] = s;
  else if (c < DIN_ + 128)   Bm_bf[row * 128 + (c - DIN_)] = f2bf(s);
  else                       Cm_bf[row * 128 + (c - DIN_ - 128)] = f2bf(s);
}

// =====================================================================
// Kernel 3: dt = softplus(dt_raw + bias); dA = dt*A; intra-chunk cumsum
// =====================================================================
__global__ __launch_bounds__(256) void dt_scan_kernel(
    const float* __restrict__ zx, const float* __restrict__ dt_bias,
    const float* __restrict__ A_log, float* __restrict__ dt_out,
    float* __restrict__ acum)
{
  __shared__ float sbuf[256];
  int bid = blockIdx.x;                 // b*512 + h*8 + c
  int c = bid & 7, h = (bid >> 3) & 63, b = bid >> 9;
  int l = threadIdx.x;
  int lg = c * CHUNK_ + l;
  float raw = zx[(size_t)(b * L_ + lg) * DPROJ_ + 2 * DIN_ + 256 + h] + dt_bias[h];
  float dt = (raw > 20.f) ? raw : log1pf(__expf(raw));
  dt_out[(size_t)(b * L_ + lg) * NH_ + h] = dt;
  float dA = dt * (-__expf(A_log[h]));
  sbuf[l] = dA;
  __syncthreads();
  for (int off = 1; off < 256; off <<= 1) {
    float v = (l >= off) ? sbuf[l - off] : 0.f;
    __syncthreads();
    sbuf[l] += v;
    __syncthreads();
  }
  acum[(size_t)((b * NH_ + h) * NC_ + c) * CHUNK_ + l] = sbuf[l];
}

// =====================================================================
// Kernel 4: SSD chunk core per (b,c,h):  S = C@B^T (panelized),
// G = mask(S)*exp(ac[l]-ac[s]), Y_diag += G@xd, state = (decay.*xd)^T@B
// B/C read as preconverted bf16 (direct u32 fragment loads).
// =====================================================================
__global__ __launch_bounds__(256) void chunk_ssm_kernel(
    const float* __restrict__ x,   const float* __restrict__ dt,
    const unsigned short* __restrict__ Bm_bf,
    const unsigned short* __restrict__ Cm_bf,
    const float* __restrict__ acum, float* __restrict__ Ydiag,
    float* __restrict__ states)
{
  __shared__ unsigned ldsXT[64 * 128];   // xd^T: [p][l/2] bf16 pairs, 32 KB
  __shared__ unsigned ldsS[8][512];      // per-wave G: [m(32)][s/2(16)], 16 KB
  __shared__ float ac[256];
  int bid = blockIdx.x;                  // b*512 + c*64 + h
  int h = bid & 63, c = (bid >> 6) & 7, b = bid >> 9;
  int tid = threadIdx.x, lane = tid & 31, wave = tid >> 5;
  int mlo = lane & 15, kh = lane >> 4;
  size_t rowbase = (size_t)(b * L_ + c * CHUNK_);
  const unsigned* B32 = (const unsigned*)Bm_bf;  // [row][n/2]
  const unsigned* C32 = (const unsigned*)Cm_bf;

  ac[tid] = acum[(size_t)((b * NH_ + h) * NC_ + c) * CHUNK_ + tid];
  for (int i = tid; i < 64 * 128; i += 256) {    // stage xd^T (bf16 pairs)
    int p = i >> 7, l2 = (i & 127) << 1;
    float x0 = x[(rowbase + l2)     * (size_t)DIN_ + h * HD_ + p] *
               dt[(rowbase + l2)     * NH_ + h];
    float x1 = x[(rowbase + l2 + 1) * (size_t)DIN_ + h * HD_ + p] *
               dt[(rowbase + l2 + 1) * NH_ + h];
    ldsXT[i] = pack2(x0, x1);
  }
  __syncthreads();

  v8f Y[2][4] = {};
  for (int s0 = 0; s0 < CHUNK_; s0 += 32) {
    // ---- S panel: rows = wave*32..+31, cols = s0..s0+31, K = 128 (n) ----
    v8f S[2][2] = {};
    for (int kk = 0; kk < 4; ++kk) {
      Frag af[2];
      for (int rt = 0; rt < 2; ++rt) {
        int row = wave * 32 + rt * 16 + mlo;
        for (int v = 0; v < 8; ++v) {
          int n = ((v < 4) ? 0 : 16) + kh * 8 + 2 * (v & 3);
          af[rt].u[v] = C32[(rowbase + row) * 64 + ((kk * 32 + n) >> 1)];
        }
      }
      for (int st = 0; st < 2; ++st) {
        Frag bf;
        int s = s0 + st * 16 + mlo;
        for (int v = 0; v < 8; ++v)
          bf.u[v] = B32[(rowbase + s) * 64 + kk * 16 + ((v < 4) ? 0 : 8) + kh * 4 + (v & 3)];
        S[0][st] = wmma_bf16(af[0], bf, S[0][st]);
        S[1][st] = wmma_bf16(af[1], bf, S[1][st]);
      }
    }
    // ---- mask + segsum decay, D-layout -> per-wave LDS (bf16) ----
    unsigned short* sw = (unsigned short*)ldsS[wave];
    for (int rt = 0; rt < 2; ++rt)
      for (int st = 0; st < 2; ++st)
        for (int r = 0; r < 8; ++r) {
          int ml  = rt * 16 + kh * 8 + r;          // local row in [0,32)
          int row = wave * 32 + ml;                // chunk position l
          int s   = s0 + st * 16 + mlo;
          float v = (s <= row) ? S[rt][st][r] * __expf(ac[row] - ac[s]) : 0.f;
          sw[ml * 32 + st * 16 + mlo] = f2bf(v);
        }
    __syncthreads();
    // ---- Y += G @ xd[s-panel] (K = 32) ----
    for (int rt = 0; rt < 2; ++rt) {
      Frag g;
      for (int v = 0; v < 8; ++v) {
        int k = ((v < 4) ? 0 : 16) + kh * 8 + 2 * (v & 3);
        g.u[v] = ldsS[wave][(rt * 16 + mlo) * 16 + (k >> 1)];
      }
      for (int ct = 0; ct < 4; ++ct) {
        Frag xb;
        int p = ct * 16 + mlo;
        for (int v = 0; v < 8; ++v) {
          int k = ((v < 4) ? 0 : 16) + kh * 8 + 2 * (v & 3);
          xb.u[v] = ldsXT[p * 128 + ((s0 + k) >> 1)];
        }
        Y[rt][ct] = wmma_bf16(g, xb, Y[rt][ct]);
      }
    }
    __syncthreads();
  }
  // ---- store Y_diag [b,l,h,p] ----
  for (int rt = 0; rt < 2; ++rt)
    for (int ct = 0; ct < 4; ++ct)
      for (int r = 0; r < 8; ++r) {
        int row = wave * 32 + rt * 16 + kh * 8 + r;
        int p   = ct * 16 + mlo;
        Ydiag[((rowbase + row) * NH_ + h) * HD_ + p] = Y[rt][ct][r];
      }

  // ---- chunk state: state[p][n] = sum_l decay(l)*xd[l][p]*B[l][n], K=256 ----
  int pt = wave >> 1, ntb = (wave & 1) * 4;
  v8f ST[4] = {};
  float acl = ac[255];
  for (int kk = 0; kk < 8; ++kk) {
    Frag af;
    int p = pt * 16 + mlo;
    for (int v = 0; v < 8; ++v) {
      int l = kk * 32 + ((v < 4) ? 0 : 16) + kh * 8 + 2 * (v & 3);
      unsigned pr = ldsXT[p * 128 + (l >> 1)];
      float lo = bf2f((unsigned short)pr)         * __expf(acl - ac[l]);
      float hi = bf2f((unsigned short)(pr >> 16)) * __expf(acl - ac[l + 1]);
      af.u[v] = pack2(lo, hi);
    }
    for (int j = 0; j < 4; ++j) {
      Frag bf;
      int n = (ntb + j) * 16 + mlo;
      for (int v = 0; v < 8; ++v) {
        int l = kk * 32 + ((v < 4) ? 0 : 16) + kh * 8 + 2 * (v & 3);
        bf.u[v] = (unsigned)Bm_bf[(rowbase + l) * 128 + n] |
                  ((unsigned)Bm_bf[(rowbase + l + 1) * 128 + n] << 16);
      }
      ST[j] = wmma_bf16(af, bf, ST[j]);
    }
  }
  size_t sbase = ((size_t)(b * NC_ + c) * NH_ + h);
  for (int j = 0; j < 4; ++j)
    for (int r = 0; r < 8; ++r) {
      int p = pt * 16 + kh * 8 + r;
      int n = (ntb + j) * 16 + mlo;
      states[(sbase * HD_ + p) * DSTATE_ + n] = ST[j][r];
    }
}

// =====================================================================
// Kernel 5: inter-chunk state recurrence (serial over 8 chunks)
// =====================================================================
__global__ __launch_bounds__(256) void state_recurrence_kernel(
    const float* __restrict__ states, const float* __restrict__ acum,
    float* __restrict__ prev)
{
  int bh = blockIdx.x;                   // b*64+h
  int b = bh >> 6, h = bh & 63;
  int tid = threadIdx.x;
  float carry[32];
  for (int i = 0; i < 32; ++i) carry[i] = 0.f;
  for (int c = 0; c < NC_; ++c) {
    size_t base = ((size_t)(b * NC_ + c) * NH_ + h) * (HD_ * DSTATE_);
    for (int i = 0; i < 32; ++i) prev[base + tid * 32 + i] = carry[i];
    float e = __expf(acum[(size_t)((b * NH_ + h) * NC_ + c) * CHUNK_ + 255]);
    for (int i = 0; i < 32; ++i)
      carry[i] = carry[i] * e + states[base + tid * 32 + i];
  }
}

// =====================================================================
// Kernel 6: Y_off = exp(ac[l]) * (C @ prev^T) ; y = Y_diag+Y_off+x*D
// =====================================================================
__global__ __launch_bounds__(256) void yoff_combine_kernel(
    const unsigned short* __restrict__ Cm_bf, const float* __restrict__ prev,
    const float* __restrict__ acum, const float* __restrict__ Ydiag,
    const float* __restrict__ x, const float* __restrict__ Dp,
    float* __restrict__ ybuf)
{
  __shared__ float ac[256];
  int bid = blockIdx.x;                  // b*512 + c*64 + h
  int h = bid & 63, c = (bid >> 6) & 7, b = bid >> 9;
  int tid = threadIdx.x, lane = tid & 31, wave = tid >> 5;
  int mlo = lane & 15, kh = lane >> 4;
  const unsigned* C32 = (const unsigned*)Cm_bf;
  ac[tid] = acum[(size_t)((b * NH_ + h) * NC_ + c) * CHUNK_ + tid];
  __syncthreads();
  size_t rowbase = (size_t)(b * L_ + c * CHUNK_);
  size_t pbase   = ((size_t)(b * NC_ + c) * NH_ + h) * (HD_ * DSTATE_);

  v8f Y[2][4] = {};
  for (int kk = 0; kk < 4; ++kk) {       // K = 128 (n)
    Frag af[2];
    for (int rt = 0; rt < 2; ++rt) {
      int row = wave * 32 + rt * 16 + mlo;
      for (int v = 0; v < 8; ++v) {
        int n = ((v < 4) ? 0 : 16) + kh * 8 + 2 * (v & 3);
        af[rt].u[v] = C32[(rowbase + row) * 64 + ((kk * 32 + n) >> 1)];
      }
    }
    for (int ct = 0; ct < 4; ++ct) {
      Frag bf;
      int p = ct * 16 + mlo;
      const float* prow = prev + pbase + (size_t)p * DSTATE_ + kk * 32;
      for (int v = 0; v < 8; ++v) {
        int n = ((v < 4) ? 0 : 16) + kh * 8 + 2 * (v & 3);
        bf.u[v] = pack2(prow[n], prow[n + 1]);
      }
      Y[0][ct] = wmma_bf16(af[0], bf, Y[0][ct]);
      Y[1][ct] = wmma_bf16(af[1], bf, Y[1][ct]);
    }
  }
  float Dh = Dp[h];
  for (int rt = 0; rt < 2; ++rt)
    for (int ct = 0; ct < 4; ++ct)
      for (int r = 0; r < 8; ++r) {
        int row = wave * 32 + rt * 16 + kh * 8 + r;
        int p   = ct * 16 + mlo;
        float xv = x[(rowbase + row) * (size_t)DIN_ + h * HD_ + p];
        float yv = Ydiag[((rowbase + row) * NH_ + h) * HD_ + p] +
                   __expf(ac[row]) * Y[rt][ct][r] + xv * Dh;
        ybuf[(rowbase + row) * (size_t)DIN_ + h * HD_ + p] = yv;
      }
}

// =====================================================================
// Kernel 7: yg = y*silu(z); RMS over D_INNER; emit bf16 for out-GEMM
// =====================================================================
__global__ __launch_bounds__(256) void gate_norm_kernel(
    const float* __restrict__ zx, const float* __restrict__ norm_w,
    const float* __restrict__ ybuf, unsigned* __restrict__ ybf)
{
  __shared__ float red[256];
  size_t row = blockIdx.x;               // b*L + l
  int tid = threadIdx.x;
  float vals[16], ss = 0.f;
  for (int i = 0; i < 16; ++i) {
    int cidx = tid * 16 + i;
    float z = zx[row * DPROJ_ + cidx];
    float y = ybuf[row * DIN_ + cidx];
    float yg = y * (z / (1.f + __expf(-z)));
    vals[i] = yg; ss += yg * yg;
  }
  red[tid] = ss;
  __syncthreads();
  for (int off = 128; off; off >>= 1) {
    if (tid < off) red[tid] += red[tid + off];
    __syncthreads();
  }
  float scale = rsqrtf(red[0] / (float)DIN_ + 1e-5f);
  for (int j = 0; j < 8; ++j) {
    int cidx = tid * 16 + 2 * j;
    ybf[row * (DIN_ / 2) + tid * 8 + j] =
        pack2(vals[2 * j]     * scale * norm_w[cidx],
              vals[2 * j + 1] * scale * norm_w[cidx + 1]);
  }
}

// =====================================================================
extern "C" void kernel_launch(void* const* d_in, const int* in_sizes, int n_in,
                              void* d_out, int out_size, void* d_ws, size_t ws_size,
                              hipStream_t stream) {
  const float* u       = (const float*)d_in[0];
  const float* W_in    = (const float*)d_in[1];
  const float* conv_w  = (const float*)d_in[2];
  const float* conv_b  = (const float*)d_in[3];
  const float* dt_bias = (const float*)d_in[4];
  const float* A_log   = (const float*)d_in[5];
  const float* Dp      = (const float*)d_in[6];
  const float* norm_w  = (const float*)d_in[7];
  const float* W_out   = (const float*)d_in[8];
  float* out = (float*)d_out;

  const size_t BL = (size_t)B_ * L_;                 // 4096 token rows
  float* p = (float*)d_ws;
  float* zx     = p;  p += BL * DPROJ_;              // [BL, 8512] fp32
  float* xb     = p;  p += BL * DIN_;                // [BL, 4096] fp32
  float* dtb    = p;  p += BL * NH_;                 // [BL, 64]
  float* acum   = p;  p += (size_t)B_ * NH_ * NC_ * CHUNK_;
  float* Ydiag  = p;  p += BL * DIN_;                // [BL, 4096]
  float* states = p;  p += (size_t)B_ * NC_ * NH_ * HD_ * DSTATE_;
  float* prevS  = p;  p += (size_t)B_ * NC_ * NH_ * HD_ * DSTATE_;
  float* ybuf   = p;  p += BL * DIN_;                // y before gate/norm
  unsigned* u_bf   = (unsigned*)p;  p += BL * DMODEL_ / 2;       // bf16 pairs
  unsigned* Win_bf = (unsigned*)p;  p += (size_t)DPROJ_ * DMODEL_ / 2;
  unsigned* Wout_bf= (unsigned*)p;  p += (size_t)DMODEL_ * DIN_ / 2;
  unsigned* y_bf   = (unsigned*)p;  p += BL * DIN_ / 2;
  unsigned short* Bm_bf = (unsigned short*)p;  p += BL * 128 / 2;
  unsigned short* Cm_bf = (unsigned short*)p;  p += BL * 128 / 2;

  // 0) operand conversions fp32 -> bf16
  {
    unsigned np;
    np = (unsigned)(BL * DMODEL_ / 2);
    cvt_bf16_kernel<<<(np + 255) / 256, 256, 0, stream>>>(u, u_bf, np);
    np = (unsigned)((size_t)DPROJ_ * DMODEL_ / 2);
    cvt_bf16_kernel<<<(np + 255) / 256, 256, 0, stream>>>(W_in, Win_bf, np);
    np = (unsigned)((size_t)DMODEL_ * DIN_ / 2);
    cvt_bf16_kernel<<<(np + 255) / 256, 256, 0, stream>>>(W_out, Wout_bf, np);
  }

  // 1) in-projection GEMM: zx = u @ W_in^T  (4096 x 8512, K=2048)
  gemm_bf16_wmma<<<dim3(DPROJ_ / 64, BL / 128), 256, 0, stream>>>(
      (const unsigned short*)u_bf, (const unsigned short*)Win_bf, zx,
      (int)BL, DPROJ_, DMODEL_);

  // 2) causal conv + SiLU + split (B/C straight to bf16)
  {
    int total = B_ * L_ * CONVD_;
    conv_silu_kernel<<<(total + 255) / 256, 256, 0, stream>>>(
        zx, conv_w, conv_b, xb, Bm_bf, Cm_bf);
  }

  // 3) dt softplus + intra-chunk cumsum of dA
  dt_scan_kernel<<<B_ * NH_ * NC_, 256, 0, stream>>>(zx, dt_bias, A_log, dtb, acum);

  // 4) SSD chunk core (Y_diag + chunk states)
  chunk_ssm_kernel<<<B_ * NC_ * NH_, 256, 0, stream>>>(
      xb, dtb, Bm_bf, Cm_bf, acum, Ydiag, states);

  // 5) inter-chunk recurrence
  state_recurrence_kernel<<<B_ * NH_, 256, 0, stream>>>(states, acum, prevS);

  // 6) Y_off + combine (+ x*D skip)
  yoff_combine_kernel<<<B_ * NC_ * NH_, 256, 0, stream>>>(
      Cm_bf, prevS, acum, Ydiag, xb, Dp, ybuf);

  // 7) gating + RMS norm -> bf16 activation
  gate_norm_kernel<<<(int)BL, 256, 0, stream>>>(zx, norm_w, ybuf, y_bf);

  // 8) out-projection GEMM: out = ynorm @ W_out^T  (4096 x 2048, K=4096)
  gemm_bf16_wmma<<<dim3(DMODEL_ / 64, BL / 128), 256, 0, stream>>>(
      (const unsigned short*)y_bf, (const unsigned short*)Wout_bf, out,
      (int)BL, DMODEL_, DIN_);

  (void)in_sizes; (void)n_in; (void)out_size; (void)ws_size;
}
